// MixMoe_42442866819222
// MI455X (gfx1250) — matrix operand order repaired and checked by reading, not compiled, not run on hardware
//
#include <hip/hip_runtime.h>
#include <hip/hip_bf16.h>
#include <math.h>

// ---------------------------------------------------------------------------
// MixMoe LoRA-SwiGLU for gfx1250 (MI455X), wave32 + v_wmma_f32_16x16x32_bf16.
// Top-2 folding: ew has 2 nonzeros/token, so the E-dense down projection
// collapses to ONE GEMM over t[n,f] = sum_k w_k * silu(w1_k) * w3_k.
// Round-2 fixes: defeat cross-tile hoisting of LDS coefficient tables in the
// k_up epilogue (asm-laundered row index, v4f-vectorized ds reads) so the
// kernel stays under 256 VGPRs; split F over gridDim.y for occupancy.
// ---------------------------------------------------------------------------

typedef __attribute__((ext_vector_type(16))) __bf16       v16bf;
typedef __attribute__((ext_vector_type(8)))  float        v8f;
typedef __attribute__((ext_vector_type(4)))  unsigned int v4u;
typedef __attribute__((ext_vector_type(4)))  float        v4f;

#define NTOK  2048
#define DDIM  1024
#define FDIM  4096
#define NEXP  8
#define RRANK 16
#define LSCALE 2.0f
#define FSEG  4           // F-dimension grid split for k_up

__device__ __forceinline__ unsigned short f2bf(float f) {
  unsigned u = __builtin_bit_cast(unsigned, f);
  u += 0x7FFFu + ((u >> 16) & 1u);          // round-to-nearest-even
  return (unsigned short)(u >> 16);
}
__device__ __forceinline__ unsigned pack2bf(float lo, float hi) {
  return (unsigned)f2bf(lo) | ((unsigned)f2bf(hi) << 16);
}
__device__ __forceinline__ float bf2f(unsigned short h) {
  return __builtin_bit_cast(float, ((unsigned)h) << 16);
}

struct QPair { v4u lo, hi; };
// two 16-byte chunks -> one 16 x bf16 WMMA fragment, fully in registers
__device__ __forceinline__ v16bf ld_frag(const unsigned short* p0,
                                         const unsigned short* p1) {
  QPair q{*(const v4u*)p0, *(const v4u*)p1};
  return __builtin_bit_cast(v16bf, q);
}
__device__ __forceinline__ float dot4(v4f a, v4f b) {
  return a[0] * b[0] + a[1] * b[1] + a[2] * b[2] + a[3] * b[3];
}

// --------------------- fp32 -> bf16 copy (8 elems/thread) ------------------
__global__ void k_cvt_bf16(const float* __restrict__ src,
                           unsigned short* __restrict__ dst, int n8) {
  int i = blockIdx.x * blockDim.x + threadIdx.x;
  int stride = gridDim.x * blockDim.x;
  for (; i < n8; i += stride) {
    v4f a = *(const v4f*)(src + (size_t)i * 8);
    v4f b = *(const v4f*)(src + (size_t)i * 8 + 4);
    v4u o = {pack2bf(a[0], a[1]), pack2bf(a[2], a[3]),
             pack2bf(b[0], b[1]), pack2bf(b[2], b[3])};
    *(v4u*)(dst + (size_t)i * 8) = o;
  }
}

// ------------------------------- routing -----------------------------------
__global__ void k_route(const float* __restrict__ x, const float* __restrict__ gw,
                        int* __restrict__ topi, float* __restrict__ topw) {
  int wave = (blockIdx.x * blockDim.x + threadIdx.x) >> 5;
  int lane = threadIdx.x & 31;
  if (wave >= NTOK) return;
  const float* xr = x + (size_t)wave * DDIM;
  float acc[NEXP];
#pragma unroll
  for (int e = 0; e < NEXP; ++e) acc[e] = 0.f;
  for (int d = lane; d < DDIM; d += 32) {
    float xv = xr[d];
#pragma unroll
    for (int e = 0; e < NEXP; ++e) acc[e] += xv * gw[e * DDIM + d];
  }
#pragma unroll
  for (int e = 0; e < NEXP; ++e)
    for (int m = 16; m >= 1; m >>= 1) acc[e] += __shfl_xor(acc[e], m, 32);
  if (lane == 0) {
    float mx = acc[0];
    for (int e = 1; e < NEXP; ++e) mx = fmaxf(mx, acc[e]);
    float p[NEXP], s = 0.f;
    for (int e = 0; e < NEXP; ++e) { p[e] = __expf(acc[e] - mx); s += p[e]; }
    for (int e = 0; e < NEXP; ++e) p[e] /= s;
    int i0 = 0;
    for (int e = 1; e < NEXP; ++e) if (p[e] > p[i0]) i0 = e;
    int i1 = (i0 == 0) ? 1 : 0;
    for (int e = 0; e < NEXP; ++e) if (e != i0 && p[e] > p[i1]) i1 = e;
    float w0 = p[i0], w1 = p[i1], wsum = w0 + w1;
    topi[wave * 2 + 0] = i0;  topi[wave * 2 + 1] = i1;
    topw[wave * 2 + 0] = w0 / wsum;  topw[wave * 2 + 1] = w1 / wsum;
  }
}

// ----------------- LoRA-up coefficients: a = x . A[e]^T --------------------
__global__ void k_lora_up_coef(const float* __restrict__ x,
                               const float* __restrict__ A1,
                               const float* __restrict__ A3,
                               const int* __restrict__ topi,
                               float* __restrict__ a1c, float* __restrict__ a3c) {
  int wave = (blockIdx.x * blockDim.x + threadIdx.x) >> 5;
  int lane = threadIdx.x & 31;
  if (wave >= NTOK * 2) return;
  int n = wave >> 1, slot = wave & 1;
  int e = topi[n * 2 + slot];
  const float* xr  = x  + (size_t)n * DDIM;
  const float* a1b = A1 + (size_t)e * RRANK * DDIM;
  const float* a3b = A3 + (size_t)e * RRANK * DDIM;
  float c1[RRANK], c3[RRANK];
#pragma unroll
  for (int r = 0; r < RRANK; ++r) { c1[r] = 0.f; c3[r] = 0.f; }
  for (int d = lane; d < DDIM; d += 32) {
    float xv = xr[d];
#pragma unroll
    for (int r = 0; r < RRANK; ++r) {
      c1[r] += xv * a1b[r * DDIM + d];
      c3[r] += xv * a3b[r * DDIM + d];
    }
  }
#pragma unroll
  for (int r = 0; r < RRANK; ++r)
    for (int m = 16; m >= 1; m >>= 1) {
      c1[r] += __shfl_xor(c1[r], m, 32);
      c3[r] += __shfl_xor(c3[r], m, 32);
    }
  if (lane == 0)
    for (int r = 0; r < RRANK; ++r) {
      a1c[(n * 2 + slot) * RRANK + r] = c1[r];
      a3c[(n * 2 + slot) * RRANK + r] = c3[r];
    }
}

// -------- up GEMMs + LoRA + SwiGLU + weight-folded merge (WMMA) ------------
// grid: (NTOK/16, FSEG).  block = 16 tokens, 8 waves sweep FDIM/16/FSEG = 64 tiles.
__global__ void __launch_bounds__(256)
k_up(const unsigned short* __restrict__ xb,
     const unsigned short* __restrict__ w1b, const unsigned short* __restrict__ w3b,
     const float* __restrict__ B1, const float* __restrict__ B3,
     const int* __restrict__ topi, const float* __restrict__ topw,
     const float* __restrict__ a1c, const float* __restrict__ a3c,
     unsigned short* __restrict__ tbuf,
     unsigned short* __restrict__ s0, unsigned short* __restrict__ s1) {
  __shared__ unsigned short sX[16 * DDIM];          // 32 KB bf16 x tile
  __shared__ v4f  sA1[16][2][4], sA3[16][2][4];     // rank-16 coeffs as v4f
  __shared__ float sW[16][2];
  __shared__ int   sE[16][2];

  int tid = threadIdx.x;
  int n0 = blockIdx.x * 16;
  {
    const v4u* g = (const v4u*)(xb + (size_t)n0 * DDIM);
    v4u* l = (v4u*)sX;
    for (int i = tid; i < 16 * DDIM / 8; i += 256) l[i] = g[i];
  }
  if (tid < 32) {
    int n = tid >> 1, s = tid & 1;
    sE[n][s] = topi[(n0 + n) * 2 + s];
    sW[n][s] = topw[(n0 + n) * 2 + s];
  }
  for (int i = tid; i < 16 * 2 * RRANK; i += 256) {
    int n = i / (2 * RRANK), rem = i % (2 * RRANK), s = rem / RRANK, r = rem % RRANK;
    ((float*)sA1)[(n * 2 + s) * RRANK + r] = a1c[((n0 + n) * 2 + s) * RRANK + r];
    ((float*)sA3)[(n * 2 + s) * RRANK + r] = a3c[((n0 + n) * 2 + s) * RRANK + r];
  }
  __syncthreads();

  int wave = tid >> 5, lane = tid & 31;
  int hi = lane >> 4;            // lane half selects K sub-range per ISA layout
  int l16 = lane & 15;
  const unsigned short* aBase = &sX[l16 * DDIM + (hi ? 8 : 0)];
  int tbase = blockIdx.y * (FDIM / 16 / FSEG);
  int tend  = tbase + (FDIM / 16 / FSEG);

#pragma unroll 1
  for (int tile = tbase + wave; tile < tend; tile += 8) {
    int f0 = tile * 16;
    const unsigned short* b1Base = w1b + (size_t)(f0 + l16) * DDIM + (hi ? 8 : 0);
    const unsigned short* b3Base = w3b + (size_t)(f0 + l16) * DDIM + (hi ? 8 : 0);
    v8f c1 = {}; v8f c3 = {};
#pragma unroll 2
    for (int kb = 0; kb < DDIM; kb += 32) {
      v16bf a  = ld_frag(aBase + kb, aBase + kb + 16);     // ds_load_b128 x2
      v16bf b1 = ld_frag(b1Base + kb, b1Base + kb + 16);   // global_load_b128 x2
      v16bf b3 = ld_frag(b3Base + kb, b3Base + kb + 16);
      c1 = __builtin_amdgcn_wmma_f32_16x16x32_bf16(false, a, false, b1,
                                                   (short)0, c1, false, false);
      c3 = __builtin_amdgcn_wmma_f32_16x16x32_bf16(false, a, false, b3,
                                                   (short)0, c3, false, false);
    }
    // epilogue: per-slot rank-16 LoRA add, SwiGLU, fold top-2 weights into t
    int f = f0 + l16;
#pragma unroll
    for (int i = 0; i < 8; ++i) {
      int row = i + (hi ? 8 : 0);
      // launder: force per-tile ds reads of coefficients instead of hoisting
      // 512 floats into VGPRs across the tile loop (round-2 spill source)
      asm volatile("" : "+v"(row));
      int n = n0 + row;
      float sv[2];
#pragma unroll
      for (int s = 0; s < 2; ++s) {
        int e = sE[row][s];
        const v4f* b1r = (const v4f*)(B1 + ((size_t)e * FDIM + f) * RRANK);
        const v4f* b3r = (const v4f*)(B3 + ((size_t)e * FDIM + f) * RRANK);
        float l1 = 0.f, l3 = 0.f;
#pragma unroll
        for (int j = 0; j < 4; ++j) {
          l1 += dot4(sA1[row][s][j], b1r[j]);
          l3 += dot4(sA3[row][s][j], b3r[j]);
        }
        float w1v = c1[i] + LSCALE * l1;
        float w3v = c3[i] + LSCALE * l3;
        sv[s] = w1v / (1.f + __expf(-w1v)) * w3v;        // silu(w1) * w3
      }
      s0[(size_t)n * FDIM + f] = f2bf(sv[0]);
      s1[(size_t)n * FDIM + f] = f2bf(sv[1]);
      tbuf[(size_t)n * FDIM + f] = f2bf(sW[row][0] * sv[0] + sW[row][1] * sv[1]);
    }
  }
}

// ----------------- u2[n,slot,r] = s_slot[n,:] . A2[e]^T --------------------
__global__ void k_u2(const unsigned short* __restrict__ sb0,
                     const unsigned short* __restrict__ sb1,
                     const float* __restrict__ A2, const int* __restrict__ topi,
                     float* __restrict__ u2) {
  int wave = (blockIdx.x * blockDim.x + threadIdx.x) >> 5;
  int lane = threadIdx.x & 31;
  if (wave >= NTOK * 2) return;
  int n = wave >> 1, slot = wave & 1;
  int e = topi[n * 2 + slot];
  const unsigned short* sr = (slot ? sb1 : sb0) + (size_t)n * FDIM;
  const float* a2b = A2 + (size_t)e * RRANK * FDIM;
  float acc[RRANK];
#pragma unroll
  for (int r = 0; r < RRANK; ++r) acc[r] = 0.f;
  for (int f = lane; f < FDIM; f += 32) {
    float sv = bf2f(sr[f]);
#pragma unroll
    for (int r = 0; r < RRANK; ++r) acc[r] += sv * a2b[r * FDIM + f];
  }
#pragma unroll
  for (int r = 0; r < RRANK; ++r)
    for (int m = 16; m >= 1; m >>= 1) acc[r] += __shfl_xor(acc[r], m, 32);
  if (lane == 0)
    for (int r = 0; r < RRANK; ++r) u2[(n * 2 + slot) * RRANK + r] = acc[r];
}

// --------------------- out = t @ W2^T  (pure WMMA GEMM) --------------------
__global__ void __launch_bounds__(256)
k_down(const unsigned short* __restrict__ tbuf,
       const unsigned short* __restrict__ w2b, float* __restrict__ out) {
  int gw = (blockIdx.x * blockDim.x + threadIdx.x) >> 5;
  int lane = threadIdx.x & 31;
  int tm = gw / (DDIM / 16);
  int tn = gw % (DDIM / 16);
  if (tm >= NTOK / 16) return;
  int n0 = tm * 16, d0 = tn * 16;
  int hi = lane >> 4, l16 = lane & 15;
  const unsigned short* aBase = tbuf + (size_t)(n0 + l16) * FDIM + (hi ? 8 : 0);
  const unsigned short* bBase = w2b + (size_t)(d0 + l16) * FDIM + (hi ? 8 : 0);
  v8f c = {};
#pragma unroll 4
  for (int kb = 0; kb < FDIM; kb += 32) {
    v16bf a = ld_frag(aBase + kb, aBase + kb + 16);
    v16bf b = ld_frag(bBase + kb, bBase + kb + 16);
    c = __builtin_amdgcn_wmma_f32_16x16x32_bf16(false, a, false, b,
                                                (short)0, c, false, false);
  }
#pragma unroll
  for (int i = 0; i < 8; ++i) {
    int n = n0 + i + (hi ? 8 : 0);
    out[(size_t)n * DDIM + d0 + l16] = c[i];
  }
}

// ------------- out += sum_k w_k*SCALE * (u2[n,k,:] . B2[e_k,d,:]) ----------
__global__ void k_lora_down(const float* __restrict__ u2, const float* __restrict__ B2,
                            const int* __restrict__ topi, const float* __restrict__ topw,
                            float* __restrict__ out) {
  int idx = blockIdx.x * blockDim.x + threadIdx.x;
  if (idx >= NTOK * DDIM) return;
  int n = idx / DDIM, d = idx % DDIM;
  float val = out[idx];
#pragma unroll
  for (int s = 0; s < 2; ++s) {
    int e = topi[n * 2 + s];
    float cw = topw[n * 2 + s] * LSCALE;
    const v4f* b2r = (const v4f*)(B2 + ((size_t)e * DDIM + d) * RRANK);
    const v4f* ur  = (const v4f*)(u2 + (n * 2 + s) * RRANK);
    float acc = 0.f;
#pragma unroll
    for (int j = 0; j < 4; ++j) acc += dot4(ur[j], b2r[j]);
    val += cw * acc;
  }
  out[idx] = val;
}

// ---------------------------------------------------------------------------
extern "C" void kernel_launch(void* const* d_in, const int* in_sizes, int n_in,
                              void* d_out, int out_size, void* d_ws, size_t ws_size,
                              hipStream_t stream) {
  const float* x   = (const float*)d_in[0];
  const float* W1  = (const float*)d_in[1];
  const float* W3  = (const float*)d_in[2];
  const float* W2  = (const float*)d_in[3];
  const float* gw  = (const float*)d_in[4];
  const float* A1  = (const float*)d_in[5];
  const float* B1  = (const float*)d_in[6];
  const float* A3  = (const float*)d_in[7];
  const float* B3  = (const float*)d_in[8];
  const float* A2  = (const float*)d_in[9];
  const float* B2  = (const float*)d_in[10];
  float* out = (float*)d_out;

  // workspace carve-out (~77 MB), 256 B aligned slices
  char* p = (char*)d_ws;
  auto alloc = [&](size_t bytes) {
    char* r = p;
    p += (bytes + 255) & ~(size_t)255;
    return r;
  };
  unsigned short* xb  = (unsigned short*)alloc((size_t)NTOK * DDIM * 2);
  unsigned short* w1b = (unsigned short*)alloc((size_t)FDIM * DDIM * 2);
  unsigned short* w3b = (unsigned short*)alloc((size_t)FDIM * DDIM * 2);
  unsigned short* w2b = (unsigned short*)alloc((size_t)DDIM * FDIM * 2);
  unsigned short* tb  = (unsigned short*)alloc((size_t)NTOK * FDIM * 2);
  unsigned short* s0  = (unsigned short*)alloc((size_t)NTOK * FDIM * 2);
  unsigned short* s1  = (unsigned short*)alloc((size_t)NTOK * FDIM * 2);
  int*   topi = (int*)alloc((size_t)NTOK * 2 * 4);
  float* topw = (float*)alloc((size_t)NTOK * 2 * 4);
  float* a1c  = (float*)alloc((size_t)NTOK * 2 * RRANK * 4);
  float* a3c  = (float*)alloc((size_t)NTOK * 2 * RRANK * 4);
  float* u2   = (float*)alloc((size_t)NTOK * 2 * RRANK * 4);

  k_cvt_bf16<<<256,  256, 0, stream>>>(x,  xb,  NTOK * DDIM / 8);
  k_cvt_bf16<<<1024, 256, 0, stream>>>(W1, w1b, FDIM * DDIM / 8);
  k_cvt_bf16<<<1024, 256, 0, stream>>>(W3, w3b, FDIM * DDIM / 8);
  k_cvt_bf16<<<1024, 256, 0, stream>>>(W2, w2b, DDIM * FDIM / 8);
  k_route<<<NTOK / 8, 256, 0, stream>>>(x, gw, topi, topw);
  k_lora_up_coef<<<NTOK * 2 / 8, 256, 0, stream>>>(x, A1, A3, topi, a1c, a3c);
  k_up<<<dim3(NTOK / 16, FSEG), 256, 0, stream>>>(xb, w1b, w3b, B1, B3, topi,
                                                  topw, a1c, a3c, tb, s0, s1);
  k_u2<<<NTOK * 2 / 8, 256, 0, stream>>>(s0, s1, A2, topi, u2);
  k_down<<<(NTOK / 16) * (DDIM / 16) / 8, 256, 0, stream>>>(tb, w2b, out);
  k_lora_down<<<NTOK * DDIM / 256, 256, 0, stream>>>(u2, B2, topi, topw, out);
}
